// GPRGNN_89910845374856
// MI455X (gfx1250) — compile-verified, matrix operand order checked by reading, not run on hardware
//
#include <hip/hip_runtime.h>
#include <hip/hip_bf16.h>

// ---------------------------------------------------------------------------
// GPR-GNN for MI455X (gfx1250, wave32, WMMA)
//   h0 = relu(x@W1+b1)@W2+b2          -- bf16 WMMA, f32 accumulate
//   out = sum_k gamma_k * A_hat^k h0  -- L2-resident gather/scatter, hw f32 atomics
// ---------------------------------------------------------------------------

#define GNN_IN   512
#define GNN_HID  256
#define GNN_OUT  64
#define GNN_K    10

typedef __attribute__((ext_vector_type(16))) __bf16 v16bf;
typedef __attribute__((ext_vector_type(8)))  float  v8f;

union ABPack {
    unsigned int u[8];
    v16bf v;
};

__device__ __forceinline__ unsigned short f2bf(float f) {
    unsigned int u = __builtin_bit_cast(unsigned int, f);
    // round-to-nearest-even bf16
    unsigned int r = u + 0x7FFFu + ((u >> 16) & 1u);
    return (unsigned short)(r >> 16);
}

__device__ __forceinline__ void atomic_fadd_hw(float* p, float v) {
#if defined(__HIP_DEVICE_COMPILE__)
    unsafeAtomicAdd(p, v);      // guarantees global_atomic_add_f32 on CDNA
#else
    atomicAdd(p, v);
#endif
}

// ---------------------------------------------------------------------------
// Repack a row-major fp32 [K x Ncol] weight matrix into the WMMA B-operand
// layout: wp[((kt*NT + nt)*32 + lane)*8 + v] = packed bf16 pair (k, k+1)
// where n = nt*16 + lane%16, half = lane/16,
//       k = kt*32 + (v<4 ? half*8 + 2v : 16 + half*8 + 2(v-4))
// ---------------------------------------------------------------------------
__global__ void gpr_repack_b(const float* __restrict__ W, unsigned int* __restrict__ wp,
                             int K, int Ncol) {
    int idx = blockIdx.x * blockDim.x + threadIdx.x;
    int NT = Ncol >> 4;
    int total = (K >> 5) * NT * 32 * 8;
    if (idx >= total) return;
    int v    = idx & 7;
    int lane = (idx >> 3) & 31;
    int rest = idx >> 8;
    int nt   = rest % NT;
    int kt   = rest / NT;
    int half = lane >> 4;
    int n    = nt * 16 + (lane & 15);
    int kk   = kt * 32 + ((v < 4) ? (half * 8 + v * 2) : (16 + half * 8 + (v - 4) * 2));
    unsigned short lo = f2bf(W[(size_t)kk * Ncol + n]);
    unsigned short hi = f2bf(W[(size_t)(kk + 1) * Ncol + n]);
    wp[idx] = (unsigned int)lo | ((unsigned int)hi << 16);
}

// ---------------------------------------------------------------------------
// Fused MLP: one wave per 16-row tile.
//   MLP1: h1 = relu(x@W1+b1) -> per-wave LDS (bf16, [16][256])
//   MLP2: h0 = h1@W2+b2      -> global
// ---------------------------------------------------------------------------
__global__ __launch_bounds__(256) void gpr_mlp_wmma(
    const float* __restrict__ x,
    const unsigned int* __restrict__ w1p,
    const unsigned int* __restrict__ w2p,
    const float* __restrict__ b1,
    const float* __restrict__ b2,
    float* __restrict__ h0,
    int n_nodes) {

    __shared__ unsigned short h1lds[8][16 * GNN_HID];   // 8 waves * 8KB

    const int lane  = threadIdx.x & 31;
    const int wave  = threadIdx.x >> 5;
    const int mtile = blockIdx.x * 8 + wave;
    if (mtile * 16 >= n_nodes) return;                  // wave-uniform exit

    const int m16  = lane & 15;
    const int half = lane >> 4;
    const float* xrow = x + (size_t)(mtile * 16 + m16) * GNN_IN;
    unsigned short* h1 = &h1lds[wave][0];

    // ---------------- MLP1: 16 n-tiles of 16 cols, K = 512 (16 k-steps) ----
    for (int nt = 0; nt < GNN_HID / 16; ++nt) {
        float bias = b1[nt * 16 + m16];
        v8f acc = {bias, bias, bias, bias, bias, bias, bias, bias};
        for (int kt = 0; kt < GNN_IN / 32; ++kt) {
            ABPack A;
            const float* pa = xrow + kt * 32 + half * 8;        // K = kt*32 + half*8 .. +7
            const float* pb = xrow + kt * 32 + 16 + half * 8;   // K = +16 ..
#pragma unroll
            for (int v = 0; v < 4; ++v) {
                A.u[v]     = (unsigned int)f2bf(pa[2 * v]) |
                             ((unsigned int)f2bf(pa[2 * v + 1]) << 16);
                A.u[4 + v] = (unsigned int)f2bf(pb[2 * v]) |
                             ((unsigned int)f2bf(pb[2 * v + 1]) << 16);
            }
            ABPack B;
            const unsigned int* bw =
                w1p + (((size_t)kt * (GNN_HID / 16) + nt) * 32 + lane) * 8;
#pragma unroll
            for (int v = 0; v < 8; ++v) B.u[v] = bw[v];
            acc = __builtin_amdgcn_wmma_f32_16x16x32_bf16(
                false, A.v, false, B.v, (short)0, acc, false, false);
        }
        // relu, C-layout scatter into per-wave LDS (bf16)
#pragma unroll
        for (int r = 0; r < 8; ++r) {
            int mm = (lane < 16) ? r : (r + 8);
            int nn = nt * 16 + (lane & 15);
            float vv = acc[r];
            vv = vv > 0.0f ? vv : 0.0f;
            h1[mm * GNN_HID + nn] = f2bf(vv);
        }
    }

    // ---------------- MLP2: 4 n-tiles of 16 cols, K = 256 (8 k-steps) ------
    for (int nt = 0; nt < GNN_OUT / 16; ++nt) {
        float bias = b2[nt * 16 + m16];
        v8f acc = {bias, bias, bias, bias, bias, bias, bias, bias};
        for (int kt = 0; kt < GNN_HID / 32; ++kt) {
            ABPack A;
            int base0 = kt * 32 + half * 8;
            int base1 = kt * 32 + 16 + half * 8;
            const unsigned short* hr = h1 + m16 * GNN_HID;
#pragma unroll
            for (int v = 0; v < 4; ++v) {
                A.u[v]     = (unsigned int)hr[base0 + 2 * v] |
                             ((unsigned int)hr[base0 + 2 * v + 1] << 16);
                A.u[4 + v] = (unsigned int)hr[base1 + 2 * v] |
                             ((unsigned int)hr[base1 + 2 * v + 1] << 16);
            }
            ABPack B;
            const unsigned int* bw =
                w2p + (((size_t)kt * (GNN_OUT / 16) + nt) * 32 + lane) * 8;
#pragma unroll
            for (int v = 0; v < 8; ++v) B.u[v] = bw[v];
            acc = __builtin_amdgcn_wmma_f32_16x16x32_bf16(
                false, A.v, false, B.v, (short)0, acc, false, false);
        }
#pragma unroll
        for (int r = 0; r < 8; ++r) {
            int mm = (lane < 16) ? r : (r + 8);
            int nn = nt * 16 + (lane & 15);
            h0[(size_t)(mtile * 16 + mm) * GNN_OUT + nn] = acc[r];
        }
    }
}

// ---------------------------------------------------------------------------
// Degree / normalization
// ---------------------------------------------------------------------------
__global__ void gpr_deg_init(float* __restrict__ deg, int n) {
    int i = blockIdx.x * blockDim.x + threadIdx.x;
    if (i < n) deg[i] = 1.0f;   // self loop
}

__global__ void gpr_deg_count(const int* __restrict__ cols, float* __restrict__ deg,
                              int e) {
    int i = blockIdx.x * blockDim.x + threadIdx.x;
    if (i < e) atomic_fadd_hw(&deg[cols[i]], 1.0f);
}

__global__ void gpr_dinv(float* __restrict__ deg, int n) {
    int i = blockIdx.x * blockDim.x + threadIdx.x;
    if (i < n) {
        float d = deg[i];
        deg[i] = d > 0.0f ? __frsqrt_rn(d) : 0.0f;   // in place: deg -> dinv
    }
}

// Precompute hop-invariant per-edge norm = dinv[row]*dinv[col]
__global__ void gpr_edge_norm(const int* __restrict__ rows, const int* __restrict__ cols,
                              const float* __restrict__ dinv, float* __restrict__ norm,
                              int e) {
    int i = blockIdx.x * blockDim.x + threadIdx.x;
    if (i < e) norm[i] = dinv[rows[i]] * dinv[cols[i]];
}

// ---------------------------------------------------------------------------
// Propagation: h_next[i] = h[i]*dinv[i]^2  (self loop), then edge scatter
// ---------------------------------------------------------------------------
__global__ void gpr_hop_init(const float* __restrict__ h, const float* __restrict__ dinv,
                             float* __restrict__ hn, long long n64) {
    long long i = blockIdx.x * (long long)blockDim.x + threadIdx.x;
    if (i < n64) {
        float di = dinv[i >> 6];
        hn[i] = h[i] * di * di;
    }
}

__global__ void gpr_scatter(const float* __restrict__ h,
                            const int* __restrict__ rows,
                            const int* __restrict__ cols,
                            const float* __restrict__ norm,
                            float* __restrict__ hn, long long e) {
    long long gid = blockIdx.x * (long long)blockDim.x + threadIdx.x;
    long long nwork = e << 4;                    // 16 threads / edge, 4 floats each
    if (gid >= nwork) return;
    int  ei = (int)(gid >> 4);
    int  c4 = (int)(gid & 15) << 2;
    int  r  = rows[ei];
    int  c  = cols[ei];
    float s = norm[ei];
    const float4 hv = *(const float4*)(h + (size_t)r * GNN_OUT + c4);
    float* dst = hn + (size_t)c * GNN_OUT + c4;
    atomic_fadd_hw(dst + 0, hv.x * s);
    atomic_fadd_hw(dst + 1, hv.y * s);
    atomic_fadd_hw(dst + 2, hv.z * s);
    atomic_fadd_hw(dst + 3, hv.w * s);
}

__global__ void gpr_accum(float* __restrict__ out, const float* __restrict__ hn,
                          const float* __restrict__ gammas, int k, float beta,
                          long long n64) {
    long long i = blockIdx.x * (long long)blockDim.x + threadIdx.x;
    if (i < n64) {
        float g = gammas[k];
        float o = (beta != 0.0f) ? out[i] * beta : 0.0f;
        out[i] = o + g * hn[i];
    }
}

// ---------------------------------------------------------------------------
// Orchestration
// ---------------------------------------------------------------------------
extern "C" void kernel_launch(void* const* d_in, const int* in_sizes, int n_in,
                              void* d_out, int out_size, void* d_ws, size_t ws_size,
                              hipStream_t stream) {
    const float* x      = (const float*)d_in[0];
    const int*   eidx   = (const int*)d_in[1];
    const float* W1     = (const float*)d_in[2];
    const float* b1     = (const float*)d_in[3];
    const float* W2     = (const float*)d_in[4];
    const float* b2     = (const float*)d_in[5];
    const float* gammas = (const float*)d_in[6];
    float*       out    = (float*)d_out;

    const int n = in_sizes[0] / GNN_IN;          // nodes
    const int e = in_sizes[1] / 2;               // edges
    const int* rows = eidx;
    const int* cols = eidx + e;

    // Workspace layout (256B aligned)
    char* ws = (char*)d_ws;
    size_t off = 0;
    auto alloc = [&](size_t bytes) {
        char* p = ws + off;
        off += (bytes + 255) & ~(size_t)255;
        return p;
    };
    unsigned int* w1p  = (unsigned int*)alloc((size_t)(GNN_IN / 32) * (GNN_HID / 16) * 32 * 8 * 4);
    unsigned int* w2p  = (unsigned int*)alloc((size_t)(GNN_HID / 32) * (GNN_OUT / 16) * 32 * 8 * 4);
    float*        dinv = (float*)alloc((size_t)n * 4);
    float*        enorm= (float*)alloc((size_t)e * 4);
    float*        h_a  = (float*)alloc((size_t)n * GNN_OUT * 4);
    float*        h_b  = (float*)alloc((size_t)n * GNN_OUT * 4);
    (void)ws_size;

    // 1) Repack weights to WMMA B-operand bf16 layout
    {
        int t1 = (GNN_IN / 32) * (GNN_HID / 16) * 32 * 8;
        gpr_repack_b<<<(t1 + 255) / 256, 256, 0, stream>>>(W1, w1p, GNN_IN, GNN_HID);
        int t2 = (GNN_HID / 32) * (GNN_OUT / 16) * 32 * 8;
        gpr_repack_b<<<(t2 + 255) / 256, 256, 0, stream>>>(W2, w2p, GNN_HID, GNN_OUT);
    }

    // 2) Degrees -> dinv -> hop-invariant edge norms
    gpr_deg_init<<<(n + 255) / 256, 256, 0, stream>>>(dinv, n);
    gpr_deg_count<<<(e + 255) / 256, 256, 0, stream>>>(cols, dinv, e);
    gpr_dinv<<<(n + 255) / 256, 256, 0, stream>>>(dinv, n);
    gpr_edge_norm<<<(e + 255) / 256, 256, 0, stream>>>(rows, cols, dinv, enorm, e);

    // 3) Fused MLP -> h_a
    {
        int mtiles = (n + 15) / 16;
        int blocks = (mtiles + 7) / 8;
        gpr_mlp_wmma<<<blocks, 256, 0, stream>>>(x, w1p, w2p, b1, b2, h_a, n);
    }

    // 4) out = gamma_0 * h
    const long long n64 = (long long)n * GNN_OUT;
    const int gblk = (int)((n64 + 255) / 256);
    gpr_accum<<<gblk, 256, 0, stream>>>(out, h_a, gammas, 0, 0.0f, n64);

    // 5) K hops of A_hat, accumulate gamma_k
    float* h_cur = h_a;
    float* h_nxt = h_b;
    const long long swork = (long long)e << 4;
    const int sblk = (int)((swork + 255) / 256);
    for (int k = 1; k <= GNN_K; ++k) {
        gpr_hop_init<<<gblk, 256, 0, stream>>>(h_cur, dinv, h_nxt, n64);
        gpr_scatter<<<sblk, 256, 0, stream>>>(h_cur, rows, cols, enorm, h_nxt, (long long)e);
        gpr_accum<<<gblk, 256, 0, stream>>>(out, h_nxt, gammas, k, 1.0f, n64);
        float* t = h_cur; h_cur = h_nxt; h_nxt = t;
    }
}
